// PVConv_76922864272023
// MI455X (gfx1250) — compile-verified, matrix operand order checked by reading, not run on hardware
//
#include <hip/hip_runtime.h>
#include <hip/hip_bf16.h>
#include <stdint.h>

typedef __attribute__((ext_vector_type(16))) __bf16 v16bf;
typedef __attribute__((ext_vector_type(8)))  __bf16 v8bf;
typedef __attribute__((ext_vector_type(8)))  float  v8f;

#define RR   32
#define R3   32768            // 32^3
#define BATCH 8
#define NPTS 32768
#define CH   64
#define LRELU(x) ((x) > 0.0f ? (x) : 0.1f * (x))

__device__ inline v16bf cat8(v8bf lo, v8bf hi) {
    v16bf r;
#pragma unroll
    for (int i = 0; i < 8; ++i) { r[i] = lo[i]; r[8 + i] = hi[i]; }
    return r;
}

__device__ inline int kmap(int half, int e) {
    // 16-bit A/B fragment K ordering per CDNA5 ISA 7.12.2:
    // half0 lanes: K = {0..7, 16..23}; half1 lanes: K = {8..15, 24..31}
    return half ? (e < 8 ? e + 8 : e + 16) : (e < 8 ? e : e + 8);
}

// ---------------- voxelize ----------------

__global__ __launch_bounds__(256)
void coord_means_k(const float* __restrict__ coords, float* __restrict__ means) {
    int bd = blockIdx.x;                    // 0..23 = b*3+d
    const float* p = coords + (long)bd * NPTS;
    float s = 0.f;
    for (int i = threadIdx.x; i < NPTS; i += 256) s += p[i];
    __shared__ float ls[256];
    ls[threadIdx.x] = s; __syncthreads();
    for (int st = 128; st > 0; st >>= 1) {
        if (threadIdx.x < st) ls[threadIdx.x] += ls[threadIdx.x + st];
        __syncthreads();
    }
    if (threadIdx.x == 0) means[bd] = ls[0] / (float)NPTS;
}

__global__ __launch_bounds__(256)
void voxel_scatter_k(const float* __restrict__ coords, const float* __restrict__ feats,
                     const float* __restrict__ means, float* __restrict__ sums,
                     float* __restrict__ cnt, float* __restrict__ normw) {
    long t = (long)blockIdx.x * 256 + threadIdx.x;
    if (t >= (long)BATCH * NPTS) return;
    int b = (int)(t >> 15); long n = t & (NPTS - 1);
    int vx[3];
#pragma unroll
    for (int d = 0; d < 3; ++d) {
        float c  = coords[((long)b * 3 + d) * NPTS + n];
        float nm = (c - means[b * 3 + d] + 1.0f) * 0.5f * (float)RR;
        nm = fminf(fmaxf(nm, 0.0f), (float)(RR - 1));
        normw[((long)b * 3 + d) * NPTS + n] = nm;
        vx[d] = (int)rintf(nm);            // round-half-even like jnp.round
    }
    long vox = (long)b * R3 + ((vx[0] * RR + vx[1]) * RR + vx[2]);
    atomicAdd(&cnt[vox], 1.0f);
    float* s = sums + vox * CH;
    const float* f = feats + (long)b * CH * NPTS + n;
    for (int c = 0; c < CH; ++c) atomicAdd(&s[c], f[(long)c * NPTS]);
}

__global__ __launch_bounds__(256)
void voxel_avg_k(const float* __restrict__ sums, const float* __restrict__ cnt,
                 __bf16* __restrict__ grid, long total) {
    long i = (long)blockIdx.x * 256 + threadIdx.x;
    if (i >= total) return;
    float c = cnt[i >> 6];
    grid[i] = (__bf16)(sums[i] / fmaxf(c, 1.0f));
}

// ---------------- weight packing (fragment layout) ----------------

__global__ __launch_bounds__(256)
void pack_conv_w_k(const float* __restrict__ w, __bf16* __restrict__ wp, int nfrag) {
    int gid = blockIdx.x * 256 + threadIdx.x;
    if (gid >= nfrag * 32) return;
    int f = gid >> 5, lane = gid & 31;
    int nb = f >> 3, kb = (f >> 2) & 1, og = f & 3;
    int o = og * 16 + (lane & 15), half = lane >> 4;
    __bf16* dst = wp + (long)f * 512 + lane * 16;
#pragma unroll
    for (int e = 0; e < 16; ++e) {
        int k = kb * 32 + kmap(half, e);           // input channel
        dst[e] = (__bf16)w[((long)o * CH + k) * 27 + nb];
    }
}

__global__ __launch_bounds__(256)
void pack_mlp_w_k(const float* __restrict__ w, __bf16* __restrict__ wp) {
    int gid = blockIdx.x * 256 + threadIdx.x;
    if (gid >= 8 * 32) return;
    int f = gid >> 5, lane = gid & 31;
    int kb = f >> 2, og = f & 3;
    int o = og * 16 + (lane & 15), half = lane >> 4;
    __bf16* dst = wp + (long)f * 512 + lane * 16;
#pragma unroll
    for (int e = 0; e < 16; ++e) {
        int k = kb * 32 + kmap(half, e);
        dst[e] = (__bf16)w[(long)o * CH + k];
    }
}

// ---------------- implicit-GEMM 3x3x3 conv via WMMA bf16 ----------------
// block = 256 threads (8 waves). Tile: full z-row (32 voxels) x 64 out-ch.
// 9 (x,y) neighbor rows staged with z-halo, double-buffered; each staged row
// feeds 3 dz offsets -> 48 WMMAs per block per stage, 1 barrier per stage.
__global__ __launch_bounds__(256)
void conv3d_wmma_k(const __bf16* __restrict__ in, const __bf16* __restrict__ wpack,
                   const float* __restrict__ bias, float* __restrict__ out) {
    __shared__ __bf16 row[2][34][CH];        // z index shifted by +1 (halo)
    int tile = blockIdx.x;                   // 8192 = B*32*32 rows
    int b = tile >> 10;
    int x = (tile >> 5) & 31;
    int y = tile & 31;
    int tid  = threadIdx.x;
    int wv   = tid >> 5, lane = tid & 31;
    int mt   = wv & 1;                       // m-tile: voxels mt*16..mt*16+15
    int og   = wv >> 1;                      // out-channel group (16 ch)
    int m    = lane & 15, half = lane >> 4;

    // staging roles: 34 voxels * 8 dword4-chunks = 272 slots over 256 threads
    int s_m0 = tid >> 3, s_dg0 = tid & 7;            // slot 0: all threads
    int s_m1 = (tid + 256) >> 3, s_dg1 = tid & 7;    // slot 1: tid < 16 only
    uint4 r0, r1;

    auto ldrow = [&](int j) {
        int xx = x + j / 3 - 1, yy = y + j % 3 - 1;
        bool rowok = (unsigned)xx < (unsigned)RR && (unsigned)yy < (unsigned)RR;
        const __bf16* rp = in + ((long)b * R3 + (xx * RR + yy) * RR) * CH;
        int zz0 = s_m0 - 1;
        r0 = make_uint4(0u, 0u, 0u, 0u);
        if (rowok && (unsigned)zz0 < (unsigned)RR)
            r0 = *(const uint4*)(rp + (long)zz0 * CH + s_dg0 * 8);
        r1 = make_uint4(0u, 0u, 0u, 0u);
        if (tid < 16) {
            int zz1 = s_m1 - 1;
            if (rowok && (unsigned)zz1 < (unsigned)RR)
                r1 = *(const uint4*)(rp + (long)zz1 * CH + s_dg1 * 8);
        }
    };
    auto strow = [&](int buf) {
        *(uint4*)&row[buf][s_m0][s_dg0 * 8] = r0;
        if (tid < 16) *(uint4*)&row[buf][s_m1][s_dg1 * 8] = r1;
    };

    v8f acc = {};
    ldrow(0);
    strow(0);
    __syncthreads();
    for (int j = 0; j < 9; ++j) {
        if (j < 8) ldrow(j + 1);             // prefetch next row into regs
        int buf = j & 1;
        int zbase = mt * 16 + m;             // output z for this lane's A row
#pragma unroll
        for (int dz = 0; dz < 3; ++dz) {
            int zr = zbase + dz;             // staged row index (halo shift)
            int nb = j * 3 + dz;             // (kd*3+kh)*3+kw neighbor index
#pragma unroll
            for (int kb = 0; kb < 2; ++kb) {
                int k0 = kb * 32;
                v16bf a  = cat8(*(const v8bf*)&row[buf][zr][k0 + half * 8],
                                *(const v8bf*)&row[buf][zr][k0 + 16 + half * 8]);
                v16bf bv = *(const v16bf*)(wpack +
                            (((long)(nb * 2 + kb) * 4 + og) * 512) + lane * 16);
                acc = __builtin_amdgcn_wmma_f32_16x16x32_bf16(
                    false, a, false, bv, (short)0, acc, false, false);
            }
        }
        if (j < 8) strow((j + 1) & 1);       // safe: writes the other buffer
        __syncthreads();
    }

    long base = ((long)b * R3 + (x * RR + y) * RR) + mt * 16;
    int och = og * 16 + (lane & 15);
    float bs = bias[och];
#pragma unroll
    for (int r = 0; r < 8; ++r) {
        int mm = r + (half ? 8 : 0);
        out[(base + mm) * CH + och] = acc[r] + bs;
    }
}

// ---------------- pointwise MLP via WMMA bf16 ----------------

__global__ __launch_bounds__(128)
void mlp_wmma_k(const float* __restrict__ feat, const __bf16* __restrict__ wpack,
                const float* __restrict__ bias, float* __restrict__ out) {
    __shared__ __bf16 patch[16][CH];
    int tile = blockIdx.x;                   // 16384 tiles
    int b  = tile >> 11;
    int n0 = (tile & 2047) << 4;
    int tid = threadIdx.x, og = tid >> 5, lane = tid & 31;
    int i = tid >> 1, seg = tid & 1;
    const float* src = feat + ((long)b * CH + i) * NPTS + n0 + seg * 8;
#pragma unroll
    for (int j = 0; j < 8; ++j) patch[seg * 8 + j][i] = (__bf16)src[j];
    __syncthreads();
    int m = lane & 15, half = lane >> 4;
    v8f acc = {};
#pragma unroll
    for (int kb = 0; kb < 2; ++kb) {
        int k0 = kb * 32;
        v16bf a  = cat8(*(const v8bf*)&patch[m][k0 + half * 8],
                        *(const v8bf*)&patch[m][k0 + 16 + half * 8]);
        v16bf bv = *(const v16bf*)(wpack + ((long)(kb * 4 + og) * 512) + lane * 16);
        acc = __builtin_amdgcn_wmma_f32_16x16x32_bf16(
            false, a, false, bv, (short)0, acc, false, false);
    }
    int och = og * 16 + (lane & 15);
    float bs = bias[och];
    long pbase = (long)b * NPTS + n0;
#pragma unroll
    for (int r = 0; r < 8; ++r) {
        int mm = r + (half ? 8 : 0);
        out[(pbase + mm) * CH + och] = acc[r] + bs;
    }
}

// ---------------- batch-norm ----------------

__global__ __launch_bounds__(256)
void chan_stats_k(const float* __restrict__ x, float* __restrict__ sum,
                  float* __restrict__ sumsq, int rowsPerBlock, long totalRows) {
    int c = threadIdx.x & 63, r4 = threadIdx.x >> 6;
    long row0 = (long)blockIdx.x * rowsPerBlock;
    float s = 0.f, ss = 0.f;
    for (long r = row0 + r4; r < row0 + rowsPerBlock && r < totalRows; r += 4) {
        float v = x[r * CH + c];
        s += v; ss += v * v;
    }
    __shared__ float ls[256], lq[256];
    ls[threadIdx.x] = s; lq[threadIdx.x] = ss;
    __syncthreads();
    if (r4 == 0) {
        s  = ls[c] + ls[64 + c] + ls[128 + c] + ls[192 + c];
        ss = lq[c] + lq[64 + c] + lq[128 + c] + lq[192 + c];
        atomicAdd(&sum[c], s);
        atomicAdd(&sumsq[c], ss);
    }
}

__global__ __launch_bounds__(256)
void bn_leaky_bf16_k(const float* __restrict__ x, const float* __restrict__ sum,
                     const float* __restrict__ sumsq, const float* __restrict__ g,
                     const float* __restrict__ be, __bf16* __restrict__ out,
                     long total, float invM) {
    long i = (long)blockIdx.x * 256 + threadIdx.x;
    if (i >= total) return;
    int c = (int)(i & 63);
    float mu  = sum[c] * invM;
    float var = sumsq[c] * invM - mu * mu;
    float v = (x[i] - mu) * rsqrtf(var + 1e-4f) * g[c] + be[c];
    out[i] = (__bf16)LRELU(v);
}

// BN+leaky for conv2 output: writes f32 v channel-major into d_out AND a
// voxel-major bf16 copy for the devoxelize gather (contiguous channels).
__global__ __launch_bounds__(256)
void bn_leaky_vout_k(const float* __restrict__ x, const float* __restrict__ sum,
                     const float* __restrict__ sumsq, const float* __restrict__ g,
                     const float* __restrict__ be, float* __restrict__ vout,
                     __bf16* __restrict__ vbf, long total, float invM) {
    long i = (long)blockIdx.x * 256 + threadIdx.x;
    if (i >= total) return;
    int c = (int)(i & 63);
    long gv = i >> 6;
    int b = (int)(gv >> 15);
    long vox = gv & (R3 - 1);
    float mu  = sum[c] * invM;
    float var = sumsq[c] * invM - mu * mu;
    float v = (x[i] - mu) * rsqrtf(var + 1e-4f) * g[c] + be[c];
    v = LRELU(v);
    vout[((long)(b * CH + c)) * R3 + vox] = v;   // channel-major (B,C,r^3)
    vbf[i] = (__bf16)v;                          // voxel-major (B,r^3,C) bf16
}

// ---------------- devoxelize + fuse ----------------
// One thread per point; v gathered voxel-major bf16 (128 B contiguous per
// corner), all 64 channels accumulated in registers; fused stores are
// coalesced across lanes (consecutive n, fixed c).
__global__ __launch_bounds__(256)
void fuse_devox_k(const __bf16* __restrict__ vbf, const float* __restrict__ normw,
                  const float* __restrict__ praw, const float* __restrict__ sum,
                  const float* __restrict__ sumsq, const float* __restrict__ g,
                  const float* __restrict__ be, float* __restrict__ fused, float invM) {
    long t = (long)blockIdx.x * 256 + threadIdx.x;      // b*NPTS + n
    if (t >= (long)BATCH * NPTS) return;
    int b = (int)(t >> 15); long n = t & (NPTS - 1);
    const float* nm = normw + (long)b * 3 * NPTS + n;
    float nx = nm[0], ny = nm[NPTS], nz = nm[2 * NPTS];
    int lx = (int)floorf(nx), ly = (int)floorf(ny), lz = (int)floorf(nz);
    float fx = nx - lx, fy = ny - ly, fz = nz - lz;
    int hx = min(lx + 1, RR - 1), hy = min(ly + 1, RR - 1), hz = min(lz + 1, RR - 1);

    float acc[CH];
#pragma unroll
    for (int c = 0; c < CH; ++c) acc[c] = 0.f;

    const __bf16* vb = vbf + (long)b * R3 * CH;
#pragma unroll
    for (int k = 0; k < 8; ++k) {
        int dx = (k >> 2) & 1, dy = (k >> 1) & 1, dz = k & 1;
        int ix = dx ? hx : lx, iy = dy ? hy : ly, iz = dz ? hz : lz;
        float w = (dx ? fx : 1.f - fx) * (dy ? fy : 1.f - fy) * (dz ? fz : 1.f - fz);
        const __bf16* vc = vb + (long)((ix * RR + iy) * RR + iz) * CH;
#pragma unroll
        for (int c8 = 0; c8 < 8; ++c8) {
            v8bf v = *(const v8bf*)(vc + c8 * 8);
#pragma unroll
            for (int j = 0; j < 8; ++j) acc[c8 * 8 + j] += w * (float)v[j];
        }
    }

    const float* pb = praw + t * CH;
    float* ob = fused + (long)b * CH * NPTS + n;
#pragma unroll
    for (int c = 0; c < CH; ++c) {
        float mu  = sum[c] * invM;
        float var = sumsq[c] * invM - mu * mu;
        float p = (pb[c] - mu) * rsqrtf(var + 1e-4f) * g[c] + be[c];
        ob[(long)c * NPTS] = acc[c] + LRELU(p);
    }
}

// ---------------- host ----------------

extern "C" void kernel_launch(void* const* d_in, const int* in_sizes, int n_in,
                              void* d_out, int out_size, void* d_ws, size_t ws_size,
                              hipStream_t stream) {
    const float* features = (const float*)d_in[0];
    const float* coords   = (const float*)d_in[1];
    const float* conv1_w  = (const float*)d_in[2];
    const float* conv1_b  = (const float*)d_in[3];
    const float* bn1_g    = (const float*)d_in[4];
    const float* bn1_b    = (const float*)d_in[5];
    const float* conv2_w  = (const float*)d_in[6];
    const float* conv2_b  = (const float*)d_in[7];
    const float* bn2_g    = (const float*)d_in[8];
    const float* bn2_b    = (const float*)d_in[9];
    const float* mlp_w    = (const float*)d_in[10];
    const float* mlp_b    = (const float*)d_in[11];
    const float* bnp_g    = (const float*)d_in[12];
    const float* bnp_b    = (const float*)d_in[13];
    float* out = (float*)d_out;

    const long nvox = (long)BATCH * R3;            // 262144
    const long nelm = nvox * CH;                   // 16,777,216
    const float invM = 1.0f / (float)nvox;

    // workspace carve-up
    uint8_t* ws = (uint8_t*)d_ws;
    size_t off = 0;
    auto carve = [&](size_t bytes) { size_t r = off; off = (off + bytes + 255) & ~(size_t)255; return r; };
    float*  sums  = (float*)(ws + carve(nelm * 4));              // 67 MB scatter sums
    float*  cnt   = (float*)(ws + carve(nvox * 4));              // 1 MB
    float*  normw = (float*)(ws + carve((long)BATCH * 3 * NPTS * 4));
    float*  means = (float*)(ws + carve(256));
    __bf16* actA  = (__bf16*)(ws + carve(nelm * 2));             // grid / act1 / v bf16
    float*  rawB  = (float*)(ws + carve(nelm * 4));              // conv raw / p raw
    __bf16* wp1   = (__bf16*)(ws + carve(27 * 8 * 512 * 2));
    __bf16* wp2   = (__bf16*)(ws + carve(27 * 8 * 512 * 2));
    __bf16* wpm   = (__bf16*)(ws + carve(8 * 512 * 2));
    float*  st1   = (float*)(ws + carve(512));                   // sum[64], sumsq[64]
    float*  st2   = (float*)(ws + carve(512));
    float*  stp   = (float*)(ws + carve(512));

    float* vout  = out + nelm;                    // second output region (B,C,r^3)
    float* fused = out;                           // first output region (B,C,N)

    // zero accumulators (capturable)
    hipMemsetAsync(sums, 0, nelm * 4, stream);
    hipMemsetAsync(cnt, 0, nvox * 4, stream);
    hipMemsetAsync(st1, 0, 512, stream);
    hipMemsetAsync(st2, 0, 512, stream);
    hipMemsetAsync(stp, 0, 512, stream);

    // voxelize
    coord_means_k<<<24, 256, 0, stream>>>(coords, means);
    voxel_scatter_k<<<(BATCH * NPTS) / 256, 256, 0, stream>>>(coords, features, means,
                                                              sums, cnt, normw);
    voxel_avg_k<<<(int)(nelm / 256), 256, 0, stream>>>(sums, cnt, actA, nelm);

    // weight packing into WMMA fragment layout
    pack_conv_w_k<<<27, 256, 0, stream>>>(conv1_w, wp1, 216);
    pack_conv_w_k<<<27, 256, 0, stream>>>(conv2_w, wp2, 216);
    pack_mlp_w_k<<<1, 256, 0, stream>>>(mlp_w, wpm);

    // conv1 -> BN -> leaky (bf16 activations for conv2)
    conv3d_wmma_k<<<8192, 256, 0, stream>>>(actA, wp1, conv1_b, rawB);
    chan_stats_k<<<128, 256, 0, stream>>>(rawB, st1, st1 + 64, 2048, nvox);
    bn_leaky_bf16_k<<<(int)(nelm / 256), 256, 0, stream>>>(rawB, st1, st1 + 64,
                                                           bn1_g, bn1_b, actA, nelm, invM);

    // conv2 -> BN -> leaky: f32 v (channel-major) into d_out + bf16 voxel-major copy
    conv3d_wmma_k<<<8192, 256, 0, stream>>>(actA, wp2, conv2_b, rawB);
    chan_stats_k<<<128, 256, 0, stream>>>(rawB, st2, st2 + 64, 2048, nvox);
    bn_leaky_vout_k<<<(int)(nelm / 256), 256, 0, stream>>>(rawB, st2, st2 + 64,
                                                           bn2_g, bn2_b, vout, actA,
                                                           nelm, invM);

    // point branch MLP -> stats ; then devoxelize + BN(p) + fuse
    mlp_wmma_k<<<16384, 128, 0, stream>>>(features, wpm, mlp_b, rawB);
    chan_stats_k<<<128, 256, 0, stream>>>(rawB, stp, stp + 64, 2048, (long)BATCH * NPTS);
    fuse_devox_k<<<(BATCH * NPTS) / 256, 256, 0, stream>>>(actA, normw, rawB, stp, stp + 64,
                                                           bnp_g, bnp_b, fused, invM);
    (void)in_sizes; (void)n_in; (void)out_size; (void)ws_size;
}
